// EndpointVectorField_38311108280934
// MI455X (gfx1250) — compile-verified, matrix operand order checked by reading, not run on hardware
//
#include <hip/hip_runtime.h>
#include <math.h>

// ---------------- CDNA5 WMMA types ----------------
typedef __attribute__((ext_vector_type(16))) __bf16 v16bf;
typedef __attribute__((ext_vector_type(8)))  float  v8f;
typedef __attribute__((ext_vector_type(4)))  int    v4i_;

#define DEV __device__ __forceinline__

// Async global->LDS (CDNA5, ASYNCcnt) if the toolchain exposes it; else ds-store staging.
#if defined(__has_builtin)
#if __has_builtin(__builtin_amdgcn_global_load_async_to_lds_b128) && \
    __has_builtin(__builtin_amdgcn_s_wait_asynccnt)
#define HAVE_ASYNC_LDS 1
#endif
#endif
#ifndef HAVE_ASYNC_LDS
#define HAVE_ASYNC_LDS 0
#endif

DEV unsigned short f2bf(float f) {
    union { float f; unsigned u; } a; a.f = f;
    unsigned r = a.u + 0x7fffu + ((a.u >> 16) & 1u);
    return (unsigned short)(r >> 16);
}

// ---------------- tiny utility kernels ----------------
__global__ void k_zero_f(float* p, long long n) {
    long long i = (long long)blockIdx.x * 256 + threadIdx.x;
    if (i < n) p[i] = 0.f;
}
__global__ void k_zero_h(unsigned short* p, long long n) {
    long long i = (long long)blockIdx.x * 256 + threadIdx.x;
    if (i < n) p[i] = 0;
}
__global__ void k_copy_f(float* d, const float* s, long long n) {
    long long i = (long long)blockIdx.x * 256 + threadIdx.x;
    if (i < n) d[i] = s[i];
}
__global__ void k_axpy(float* y, const float* x, long long n) {
    long long i = (long long)blockIdx.x * 256 + threadIdx.x;
    if (i < n) y[i] += x[i];
}

// ---------------- weight prep: fp32 [fi,fo] -> bf16 [Np x Kp] (B^T, padded) ----------------
__global__ void k_prep_w(const float* __restrict__ W, int fi, int fo_total, int col0,
                         int ncols, unsigned short* __restrict__ Wt, int Kp, int Np) {
    long long t = (long long)blockIdx.x * 256 + threadIdx.x;
    if (t >= (long long)Np * Kp) return;
    int n = (int)(t / Kp), k = (int)(t % Kp);
    float v = (n < ncols && k < fi) ? W[(size_t)k * fo_total + col0 + n] : 0.f;
    Wt[t] = f2bf(v);
}

DEV float act_apply(float v, int act) {
    if (act == 1) return v / (1.f + __expf(-v));
    if (act == 2) return 1.f / (1.f + __expf(-v));
    return v;
}

// ---------------- wide GEMM (N multiple of 64): LDS-staged weight panel ----------------
// Each block: 8 waves x (16 rows each) over one 64-col weight panel staged in LDS.
// Each wave: 16x64 tile = 4 WMMA accumulators sharing one A fragment (4x A reuse).
__global__ void k_gemm_wide(const unsigned short* __restrict__ A, int lda,
                            const unsigned short* __restrict__ W, int ldw /*==K*/,
                            const float* __restrict__ bias,
                            float* __restrict__ C, int ldc,
                            unsigned short* __restrict__ Cb, int ldcb,
                            int M, int N, int K, int act, int ngroups) {
    extern __shared__ __attribute__((aligned(32))) unsigned short Bs[];
    const int tid = threadIdx.x;
    const int ng = blockIdx.x % ngroups;
    const int mg = blockIdx.x / ngroups;

    // stage contiguous 64-row weight panel (64*K bf16) into LDS
    const unsigned short* Wpanel = W + (size_t)(ng * 64) * ldw;
    const int chunks = (64 * K) >> 3;       // 16-byte chunks
#if HAVE_ASYNC_LDS
    for (int c = tid; c < chunks; c += 256)
        __builtin_amdgcn_global_load_async_to_lds_b128(
            (v4i_*)(Wpanel + (size_t)c * 8),
            (v4i_*)(Bs + (size_t)c * 8), 0, 0);
    __builtin_amdgcn_s_wait_asynccnt(0);
#else
    for (int c = tid; c < chunks; c += 256)
        *(uint4*)(Bs + (size_t)c * 8) = *(const uint4*)(Wpanel + (size_t)c * 8);
#endif
    __syncthreads();

    const int wid = tid >> 5, lane = tid & 31;
    const int half = lane >> 4, lr = lane & 15;
    const int mt = mg * 8 + wid;
    if (mt >= (M >> 4)) return;             // wave-uniform; after the only barrier

    const unsigned short* arow = A + (size_t)(mt * 16 + lr) * lda + half * 16;
    const unsigned short* b0 = Bs + (size_t)lr * K + half * 16;

    v8f a0 = {}, a1 = {}, a2 = {}, a3 = {};
    for (int k0 = 0; k0 < K; k0 += 32) {
        v16bf a  = *(const v16bf*)(arow + k0);
        v16bf w0 = *(const v16bf*)(b0 + k0);
        v16bf w1 = *(const v16bf*)(b0 + 16 * K + k0);
        v16bf w2 = *(const v16bf*)(b0 + 32 * K + k0);
        v16bf w3 = *(const v16bf*)(b0 + 48 * K + k0);
        a0 = __builtin_amdgcn_wmma_f32_16x16x32_bf16(false, a, false, w0, (short)0, a0, false, false);
        a1 = __builtin_amdgcn_wmma_f32_16x16x32_bf16(false, a, false, w1, (short)0, a1, false, false);
        a2 = __builtin_amdgcn_wmma_f32_16x16x32_bf16(false, a, false, w2, (short)0, a2, false, false);
        a3 = __builtin_amdgcn_wmma_f32_16x16x32_bf16(false, a, false, w3, (short)0, a3, false, false);
    }
#pragma unroll
    for (int t = 0; t < 4; ++t) {
        v8f acc = (t == 0) ? a0 : (t == 1) ? a1 : (t == 2) ? a2 : a3;
        int c = ng * 64 + t * 16 + lr;
        if (c < N) {
            float bv = bias ? bias[c] : 0.f;
#pragma unroll
            for (int i = 0; i < 8; ++i) {
                int r = mt * 16 + half * 8 + i;
                float v = act_apply(acc[i] + bv, act);
                if (C)  C[(size_t)r * ldc + c] = v;
                if (Cb) Cb[(size_t)r * ldcb + c] = f2bf(v);
            }
        }
    }
}

// ---------------- small-N GEMM: one 16x16 tile per wave (gates / heads) ----------------
__global__ void k_gemm_bf16(const unsigned short* __restrict__ A, int lda,
                            const unsigned short* __restrict__ W, int ldw,
                            const float* __restrict__ bias,
                            float* __restrict__ C, int ldc,
                            unsigned short* __restrict__ Cb, int ldcb,
                            int M, int N, int K, int act) {
    int wid  = threadIdx.x >> 5;
    int lane = threadIdx.x & 31;
    int mt_tiles = M >> 4;
    int nt_tiles = (N + 15) >> 4;
    int tile = blockIdx.x * (blockDim.x >> 5) + wid;
    if (tile >= mt_tiles * nt_tiles) return;        // uniform per wave: EXEC stays all-ones
    int mt = tile % mt_tiles, nt = tile / mt_tiles;
    int half = lane >> 4, lr = lane & 15;

    const unsigned short* arow = A + (size_t)(mt * 16 + lr) * lda + half * 16;
    const unsigned short* brow = W + (size_t)(nt * 16 + lr) * ldw + half * 16;

    v8f acc = {};
    for (int k0 = 0; k0 < K; k0 += 32) {
        v16bf a = *(const v16bf*)(arow + k0);
        v16bf b = *(const v16bf*)(brow + k0);
        acc = __builtin_amdgcn_wmma_f32_16x16x32_bf16(false, a, false, b,
                                                      (short)0, acc, false, false);
    }
#pragma unroll
    for (int i = 0; i < 8; ++i) {
        int r = mt * 16 + half * 8 + i;
        int c = nt * 16 + lr;
        if (c < N) {
            float v = act_apply(acc[i] + (bias ? bias[c] : 0.f), act);
            if (C)  C[(size_t)r * ldc + c] = v;
            if (Cb) Cb[(size_t)r * ldcb + c] = f2bf(v);
        }
    }
}

// ---------------- embeddings input builders ----------------
__global__ void k_build_s_in(const float* __restrict__ a, const float* __restrict__ c,
                             const float* __restrict__ t, const int* __restrict__ nbi,
                             unsigned short* __restrict__ out, int n) {
    int i = blockIdx.x * 256 + threadIdx.x;
    if (i >= n) return;
    unsigned short* o = out + (size_t)i * 32;
    for (int j = 0; j < 10; ++j) o[j] = f2bf(a[(size_t)i * 10 + j]);
    o[10] = f2bf(t[nbi[i]]);
    for (int j = 0; j < 6; ++j) o[11 + j] = f2bf(c[(size_t)i * 6 + j]);
}
__global__ void k_build_e_in(const float* __restrict__ e, unsigned short* __restrict__ out, int E) {
    int i = blockIdx.x * 256 + threadIdx.x;
    if (i >= E) return;
    unsigned short* o = out + (size_t)i * 32;
    for (int j = 0; j < 5; ++j) o[j] = f2bf(e[(size_t)i * 5 + j]);
}

// ---------------- distances + RBF (writes bf16 RBF straight into the msg concat buffer) ----
__global__ void k_dist(const float* __restrict__ x, const int* __restrict__ src,
                       const int* __restrict__ dst, float* __restrict__ xd,
                       unsigned short* __restrict__ dout, int ldd, int doff, int E) {
    int e = blockIdx.x * 256 + threadIdx.x;
    if (e >= E) return;
    int s = src[e], d = dst[e];
    float dx = x[(size_t)s * 3 + 0] - x[(size_t)d * 3 + 0];
    float dy = x[(size_t)s * 3 + 1] - x[(size_t)d * 3 + 1];
    float dz = x[(size_t)s * 3 + 2] - x[(size_t)d * 3 + 2];
    float dij = sqrtf(dx * dx + dy * dy + dz * dz + 1e-8f) + 1e-8f;
    float inv = 1.f / dij;
    xd[(size_t)e * 3 + 0] = dx * inv;
    xd[(size_t)e * 3 + 1] = dy * inv;
    xd[(size_t)e * 3 + 2] = dz * inv;
    unsigned short* o = dout + (size_t)e * ldd + doff;
    for (int m = 0; m < 16; ++m) {
        float mu = (20.f / 15.f) * m;
        float u  = (dij - mu) * (1.f / 1.25f);
        o[m] = f2bf(__expf(-u * u));
    }
}

// ---------------- gathers ----------------
__global__ void k_gather_msg(const unsigned short* __restrict__ sbf,
                             const unsigned short* __restrict__ efbf,
                             const int* __restrict__ src,
                             unsigned short* __restrict__ out, int E) {
    long long t = (long long)blockIdx.x * 256 + threadIdx.x;
    if (t >= (long long)E * 256) return;
    int e = (int)(t >> 8), j = (int)(t & 255);
    if (j == 0 && e + 64 < E)
        __builtin_prefetch(sbf + (size_t)src[e + 64] * 160, 0, 1);  // global_prefetch_b8
    if (j < 128) out[(size_t)e * 320 + j]      = sbf[(size_t)src[e] * 160 + j];
    else         out[(size_t)e * 320 + 16 + j] = efbf[(size_t)e * 128 + (j - 128)];
}
__global__ void k_build_mv(const float* __restrict__ xdiff, const float* __restrict__ V,
                           const int* __restrict__ src, float* __restrict__ mv, int E) {
    long long t = (long long)blockIdx.x * 256 + threadIdx.x;
    if (t >= (long long)E * 17) return;
    int e = (int)(t / 17), sl = (int)(t % 17);
    float* o = mv + (size_t)e * 51 + sl * 3;
    if (sl == 0) {
        o[0] = xdiff[(size_t)e * 3]; o[1] = xdiff[(size_t)e * 3 + 1]; o[2] = xdiff[(size_t)e * 3 + 2];
    } else {
        const float* v = V + (size_t)src[e] * 48 + (sl - 1) * 3;
        o[0] = v[0]; o[1] = v[1]; o[2] = v[2];
    }
}
__global__ void k_gather_eupd(const unsigned short* __restrict__ sbf,
                              const unsigned short* __restrict__ efbf,
                              const int* __restrict__ src, const int* __restrict__ dst,
                              unsigned short* __restrict__ out, int E) {
    long long t = (long long)blockIdx.x * 256 + threadIdx.x;
    if (t >= (long long)E * 384) return;
    int e = (int)(t / 384), j = (int)(t % 384);
    if (j == 0 && e + 64 < E)
        __builtin_prefetch(sbf + (size_t)src[e + 64] * 160, 0, 1);
    unsigned short v;
    if (j < 128)      v = sbf[(size_t)src[e] * 160 + j];
    else if (j < 256) v = sbf[(size_t)dst[e] * 160 + (j - 128)];
    else              v = efbf[(size_t)e * 128 + (j - 256)];
    out[(size_t)e * 384 + j] = v;
}
__global__ void k_add_halves(const float* __restrict__ ef, unsigned short* __restrict__ out, int EH) {
    long long t = (long long)blockIdx.x * 256 + threadIdx.x;
    if (t >= (long long)EH * 128) return;
    int e = (int)(t >> 7), j = (int)(t & 127);
    out[t] = f2bf(ef[(size_t)e * 128 + j] + ef[(size_t)(e + EH) * 128 + j]);
}

// ---------------- fused GVP vector path: Vh = V.Wh, sh = |Vh|, Vu = Vh.Wu ----------------
__global__ void k_gvp_vec(const float* __restrict__ Vin, int vi, int sVin,
                          const float* __restrict__ Wh, int dh,
                          const float* __restrict__ Wu, int vo,
                          float* __restrict__ Vu,
                          unsigned short* __restrict__ sh_out, int ld_sh, int sh_off, int n) {
    int i = blockIdx.x * 256 + threadIdx.x;
    if (i >= n) return;
    float Vl[51], Vh[51];
    for (int v = 0; v < vi; ++v)
        for (int c = 0; c < 3; ++c) Vl[v * 3 + c] = Vin[(size_t)i * sVin + v * 3 + c];
    for (int h = 0; h < dh; ++h) {
        float s0 = 0, s1 = 0, s2 = 0;
        for (int v = 0; v < vi; ++v) {
            float w = Wh[(size_t)v * dh + h];
            s0 += Vl[v * 3] * w; s1 += Vl[v * 3 + 1] * w; s2 += Vl[v * 3 + 2] * w;
        }
        Vh[h * 3] = s0; Vh[h * 3 + 1] = s1; Vh[h * 3 + 2] = s2;
        sh_out[(size_t)i * ld_sh + sh_off + h] = f2bf(sqrtf(s0 * s0 + s1 * s1 + s2 * s2 + 1e-8f));
    }
    for (int u = 0; u < vo; ++u) {
        float s0 = 0, s1 = 0, s2 = 0;
        for (int h = 0; h < dh; ++h) {
            float w = Wu[(size_t)h * vo + u];
            s0 += Vh[h * 3] * w; s1 += Vh[h * 3 + 1] * w; s2 += Vh[h * 3 + 2] * w;
        }
        Vu[(size_t)i * (vo * 3) + u * 3 + 0] = s0;
        Vu[(size_t)i * (vo * 3) + u * 3 + 1] = s1;
        Vu[(size_t)i * (vo * 3) + u * 3 + 2] = s2;
    }
}

__global__ void k_gate_mul(const float* __restrict__ g, int ldg, const float* __restrict__ Vu,
                           float* __restrict__ Vo, int sOut, int vo, int n) {
    long long t = (long long)blockIdx.x * 256 + threadIdx.x;
    if (t >= (long long)n * vo) return;
    int i = (int)(t / vo), u = (int)(t % vo);
    float gv = g[(size_t)i * ldg + u];
    for (int c = 0; c < 3; ++c)
        Vo[(size_t)i * sOut + u * 3 + c] = gv * Vu[(size_t)i * (vo * 3) + u * 3 + c];
}

// ---------------- segment-sum via global f32 atomics ----------------
__global__ void k_scatter_add(const float* __restrict__ src, const int* __restrict__ idx,
                              float* __restrict__ acc, int rows, int width, float scale) {
    long long t = (long long)blockIdx.x * 256 + threadIdx.x;
    if (t >= (long long)rows * width) return;
    int r = (int)(t / width), c = (int)(t % width);
    atomicAdd(&acc[(size_t)idx[r] * width + c], src[t] * scale);
}

// ---------------- residual + LayerNorm(d=128), wave32 shfl reductions ----------------
__global__ void k_resid_ln(const float* __restrict__ xin, const float* __restrict__ add,
                           const float* __restrict__ g, const float* __restrict__ b,
                           float* __restrict__ yout, unsigned short* __restrict__ ybf,
                           int ldbf, int n) {
    int row  = blockIdx.x * 8 + (threadIdx.x >> 5);
    int lane = threadIdx.x & 31;
    if (row >= n) return;
    float v[4];
#pragma unroll
    for (int j = 0; j < 4; ++j) {
        int c = lane + j * 32;
        float x = xin[(size_t)row * 128 + c];
        if (add) x += add[(size_t)row * 128 + c];
        v[j] = x;
    }
    float tot = v[0] + v[1] + v[2] + v[3];
    for (int m = 16; m > 0; m >>= 1) tot += __shfl_xor(tot, m, 32);
    float mean = tot * (1.f / 128.f);
    float vs = 0;
#pragma unroll
    for (int j = 0; j < 4; ++j) { float d = v[j] - mean; vs += d * d; }
    for (int m = 16; m > 0; m >>= 1) vs += __shfl_xor(vs, m, 32);
    float inv = rsqrtf(vs * (1.f / 128.f) + 1e-5f);
#pragma unroll
    for (int j = 0; j < 4; ++j) {
        int c = lane + j * 32;
        float y = (v[j] - mean) * inv * g[c] + b[c];
        yout[(size_t)row * 128 + c] = y;
        if (ybf) ybf[(size_t)row * ldbf + c] = f2bf(y);
    }
}

// =====================================================================================
extern "C" void kernel_launch(void* const* d_in, const int* in_sizes, int n_in,
                              void* d_out, int out_size, void* d_ws, size_t ws_size,
                              hipStream_t stream) {
    (void)in_sizes; (void)n_in; (void)out_size; (void)ws_size;
    const int N = 10000, E = 160000, EH = 80000;
    auto F = [&](int i) { return (const float*)d_in[i]; };
    auto I = [&](int i) { return (const int*)d_in[i]; };
    auto cdv = [](long long a) { return (int)((a + 255) / 256); };

    // bump allocator over d_ws
    size_t off = 0;
    auto alloc  = [&](size_t bytes) -> void* {
        off = (off + 255) & ~(size_t)255;
        void* p = (char*)d_ws + off; off += bytes; return p;
    };
    auto allocF = [&](size_t n) { return (float*)alloc(n * 4); };
    auto allocH = [&](size_t n) { return (unsigned short*)alloc(n * 2); };

    // ---- weight prep (bf16, transposed, padded) ----
    struct WB { const unsigned short* w; int kp; };
    auto prep_cols = [&](int idx, int fi, int fo_total, int col0, int nc) -> WB {
        int Kp = ((fi + 31) / 32) * 32, Np = ((nc + 15) / 16) * 16;
        unsigned short* buf = allocH((size_t)Np * Kp);
        k_prep_w<<<cdv((long long)Np * Kp), 256, 0, stream>>>(F(idx), fi, fo_total, col0, nc, buf, Kp, Np);
        return {buf, Kp};
    };
    auto prep = [&](int idx, int fi, int fo) { return prep_cols(idx, fi, fo, 0, fo); };

    struct GW { const float *Wh, *Wu, *gb, *lb; WB gW, lW; };
    auto mkgvp = [&](int base, int fi, int fo, int vi, int vo) -> GW {
        int dh = vi > vo ? vi : vo;
        GW g; g.Wh = F(base + 0); g.Wu = F(base + 1);
        g.gW = prep(base + 2, fo, vo);      g.gb = F(base + 3);
        g.lW = prep(base + 4, dh + fi, fo); g.lb = F(base + 5);
        return g;
    };

    // JAX sorted-key pytree flattening of `params`
    GW cmsg[4][3], cupd[4][3];
    const float *lnm_b[4], *lnm_g[4], *lnu_b[4], *lnu_g[4];
    for (int ci = 0; ci < 4; ++ci) {
        int base = ci * 40;
        lnm_b[ci] = F(base + 0); lnm_g[ci] = F(base + 1);
        lnu_b[ci] = F(base + 2); lnu_g[ci] = F(base + 3);
        cmsg[ci][0] = mkgvp(base + 4, 272, 128, 17, 16);
        cmsg[ci][1] = mkgvp(base + 10, 128, 128, 16, 16);
        cmsg[ci][2] = mkgvp(base + 16, 128, 128, 16, 16);
        for (int j = 0; j < 3; ++j) cupd[ci][j] = mkgvp(base + 22 + j * 6, 128, 128, 16, 16);
    }
    WB Wee1 = prep(160, 5, 128);    const float* bee1 = F(161);
    WB Wee2 = prep(162, 128, 128);  const float* bee2 = F(163);
    const float *ee_lnb = F(164), *ee_lng = F(165);
    WB Weh1 = prep(166, 128, 128);  const float* beh1 = F(167);
    WB Weh2 = prep(168, 128, 5);    const float* beh2 = F(169);
    WB Weu1 = prep(170, 384, 128);  const float* beu1 = F(171);
    WB Weu2 = prep(172, 128, 128);  const float* beu2 = F(173);
    const float *eu_lnb = F(174), *eu_lng = F(175);
    WB Wnh1  = prep(176, 128, 128); const float* bnh1 = F(177);
    WB Wnh2a = prep_cols(178, 128, 16, 0, 10);
    WB Wnh2c = prep_cols(178, 128, 16, 10, 6);
    const float* bnh2 = F(179);
    GW pos[3];
    pos[0] = mkgvp(180, 128, 128, 16, 16);
    pos[1] = mkgvp(186, 128, 128, 16, 16);
    pos[2] = mkgvp(192, 128, 128, 16, 1);
    WB Wse1 = prep(198, 17, 128);   const float* bse1 = F(199);
    WB Wse2 = prep(200, 128, 128);  const float* bse2 = F(201);
    const float *se_lnb = F(202), *se_lng = F(203);
    // inputs: a_t=204 c_t=205 x_t=206 e_t=207 t=208 nbi=209 src=210 dst=211

    // ---- activations / scratch ----
    unsigned short* sbf  = allocH((size_t)N * 160);
    float* s    = allocF((size_t)N * 128);
    float* V    = allocF((size_t)N * 48);
    float* xbuf = allocF((size_t)N * 3);
    float* ef   = allocF((size_t)E * 128);
    unsigned short* efbf = allocH((size_t)E * 128);
    float* xdiff = allocF((size_t)E * 3);
    unsigned short* lin0  = allocH((size_t)E * 320);
    unsigned short* mbufA = allocH((size_t)E * 160);
    unsigned short* mbufB = allocH((size_t)E * 160);
    float* mv   = allocF((size_t)E * 51);
    float* vu   = allocF((size_t)E * 48);
    float* gate = allocF((size_t)E * 16);
    float* msout = allocF((size_t)E * 128);
    float* sacc = allocF((size_t)N * 128);
    float* vacc = allocF((size_t)N * 48);
    unsigned short* ubufB = allocH((size_t)N * 160);
    unsigned short* ubufC = allocH((size_t)N * 160);
    float* nvu  = allocF((size_t)N * 48);
    float* nmv  = allocF((size_t)N * 48);
    float* us   = allocF((size_t)N * 128);
    float* xupd = allocF((size_t)N * 3);
    unsigned short* h_in = allocH((size_t)E * 384);
    unsigned short* hbf  = allocH((size_t)E * 128);
    float* ef_h = allocF((size_t)E * 128);
    unsigned short* in0  = allocH((size_t)N * 32);
    unsigned short* ein  = allocH((size_t)E * 32);
    unsigned short* nhbf = allocH((size_t)N * 128);
    unsigned short* aebf = allocH((size_t)EH * 128);
    unsigned short* ehbf = allocH((size_t)EH * 128);

    auto zF = [&](float* p, long long n) { k_zero_f<<<cdv(n), 256, 0, stream>>>(p, n); };
    auto zH = [&](unsigned short* p, long long n) { k_zero_h<<<cdv(n), 256, 0, stream>>>(p, n); };
    // zero padded-concat buffers once; pads are never written afterwards
    zH(lin0, (long long)E * 320); zH(mbufA, (long long)E * 160); zH(mbufB, (long long)E * 160);
    zH(sbf, (long long)N * 160);  zH(ubufB, (long long)N * 160); zH(ubufC, (long long)N * 160);
    zH(in0, (long long)N * 32);   zH(ein, (long long)E * 32);

    auto gemm = [&](const unsigned short* A, int lda, WB W, const float* bias,
                    float* C, int ldc, unsigned short* Cb, int ldcb, int M, int Nn, int act) {
        if ((Nn & 63) == 0) {
            int ngroups = Nn / 64;
            int grid = ((M / 16 + 7) / 8) * ngroups;
            size_t shmem = (size_t)64 * W.kp * 2;
            k_gemm_wide<<<grid, 256, shmem, stream>>>(A, lda, W.w, W.kp, bias,
                                                      C, ldc, Cb, ldcb, M, Nn, W.kp, act, ngroups);
        } else {
            int tiles = (M / 16) * ((Nn + 15) / 16);
            k_gemm_bf16<<<(tiles + 7) / 8, 256, 0, stream>>>(A, lda, W.w, W.kp, bias,
                                                             C, ldc, Cb, ldcb, M, Nn, W.kp, act);
        }
    };

    // ---- scalar / edge embeddings ----
    k_build_s_in<<<cdv(N), 256, 0, stream>>>(F(204), F(205), F(208), I(209), in0, N);
    gemm(in0, 32, Wse1, bse1, nullptr, 0, nhbf, 128, N, 128, 1);
    gemm(nhbf, 128, Wse2, bse2, us, 128, nullptr, 0, N, 128, 1);
    k_resid_ln<<<(N + 7) / 8, 256, 0, stream>>>(us, nullptr, se_lng, se_lnb, s, sbf, 160, N);

    k_build_e_in<<<cdv(E), 256, 0, stream>>>(F(207), ein, E);
    gemm(ein, 32, Wee1, bee1, nullptr, 0, hbf, 128, E, 128, 1);
    gemm(hbf, 128, Wee2, bee2, ef_h, 128, nullptr, 0, E, 128, 1);
    k_resid_ln<<<(E + 7) / 8, 256, 0, stream>>>(ef_h, nullptr, ee_lng, ee_lnb, ef, efbf, 128, E);

    zF(V, (long long)N * 48);
    k_copy_f<<<cdv((long long)N * 3), 256, 0, stream>>>(xbuf, F(206), (long long)N * 3);
    k_dist<<<cdv(E), 256, 0, stream>>>(xbuf, I(210), I(211), xdiff, lin0, 320, 128, E);

    // ---- conv rounds ----
    for (int ci = 0; ci < 4; ++ci) {
        k_gather_msg<<<cdv((long long)E * 256), 256, 0, stream>>>(sbf, efbf, I(210), lin0, E);
        k_build_mv<<<cdv((long long)E * 17), 256, 0, stream>>>(xdiff, V, I(210), mv, E);

        GW g = cmsg[ci][0];
        k_gvp_vec<<<cdv(E), 256, 0, stream>>>(mv, 17, 51, g.Wh, 17, g.Wu, 16, vu, lin0, 320, 272, E);
        gemm(lin0, 320, g.lW, g.lb, nullptr, 0, mbufA, 160, E, 128, 1);
        gemm(mbufA, 160, g.gW, g.gb, gate, 16, nullptr, 0, E, 16, 2);
        k_gate_mul<<<cdv((long long)E * 16), 256, 0, stream>>>(gate, 16, vu, mv, 48, 16, E);

        g = cmsg[ci][1];
        k_gvp_vec<<<cdv(E), 256, 0, stream>>>(mv, 16, 48, g.Wh, 16, g.Wu, 16, vu, mbufA, 160, 128, E);
        gemm(mbufA, 160, g.lW, g.lb, nullptr, 0, mbufB, 160, E, 128, 1);
        gemm(mbufB, 160, g.gW, g.gb, gate, 16, nullptr, 0, E, 16, 2);
        k_gate_mul<<<cdv((long long)E * 16), 256, 0, stream>>>(gate, 16, vu, mv, 48, 16, E);

        g = cmsg[ci][2];
        k_gvp_vec<<<cdv(E), 256, 0, stream>>>(mv, 16, 48, g.Wh, 16, g.Wu, 16, vu, mbufB, 160, 128, E);
        gemm(mbufB, 160, g.lW, g.lb, msout, 128, mbufA, 160, E, 128, 1);
        gemm(mbufA, 160, g.gW, g.gb, gate, 16, nullptr, 0, E, 16, 2);
        k_gate_mul<<<cdv((long long)E * 16), 256, 0, stream>>>(gate, 16, vu, mv, 48, 16, E);

        zF(sacc, (long long)N * 128); zF(vacc, (long long)N * 48);
        k_scatter_add<<<cdv((long long)E * 128), 256, 0, stream>>>(msout, I(211), sacc, E, 128, 0.01f);
        k_scatter_add<<<cdv((long long)E * 48), 256, 0, stream>>>(mv, I(211), vacc, E, 48, 0.01f);
        k_resid_ln<<<(N + 7) / 8, 256, 0, stream>>>(s, sacc, lnm_g[ci], lnm_b[ci], s, sbf, 160, N);
        k_axpy<<<cdv((long long)N * 48), 256, 0, stream>>>(V, vacc, (long long)N * 48);

        // node update chain
        g = cupd[ci][0];
        k_gvp_vec<<<cdv(N), 256, 0, stream>>>(V, 16, 48, g.Wh, 16, g.Wu, 16, nvu, sbf, 160, 128, N);
        gemm(sbf, 160, g.lW, g.lb, nullptr, 0, ubufB, 160, N, 128, 1);
        gemm(ubufB, 160, g.gW, g.gb, gate, 16, nullptr, 0, N, 16, 2);
        k_gate_mul<<<cdv((long long)N * 16), 256, 0, stream>>>(gate, 16, nvu, nmv, 48, 16, N);
        g = cupd[ci][1];
        k_gvp_vec<<<cdv(N), 256, 0, stream>>>(nmv, 16, 48, g.Wh, 16, g.Wu, 16, nvu, ubufB, 160, 128, N);
        gemm(ubufB, 160, g.lW, g.lb, nullptr, 0, ubufC, 160, N, 128, 1);
        gemm(ubufC, 160, g.gW, g.gb, gate, 16, nullptr, 0, N, 16, 2);
        k_gate_mul<<<cdv((long long)N * 16), 256, 0, stream>>>(gate, 16, nvu, nmv, 48, 16, N);
        g = cupd[ci][2];
        k_gvp_vec<<<cdv(N), 256, 0, stream>>>(nmv, 16, 48, g.Wh, 16, g.Wu, 16, nvu, ubufC, 160, 128, N);
        gemm(ubufC, 160, g.lW, g.lb, us, 128, ubufB, 160, N, 128, 1);
        gemm(ubufB, 160, g.gW, g.gb, gate, 16, nullptr, 0, N, 16, 2);
        k_gate_mul<<<cdv((long long)N * 16), 256, 0, stream>>>(gate, 16, nvu, nmv, 48, 16, N);
        k_resid_ln<<<(N + 7) / 8, 256, 0, stream>>>(s, us, lnu_g[ci], lnu_b[ci], s, sbf, 160, N);
        k_axpy<<<cdv((long long)N * 48), 256, 0, stream>>>(V, nmv, (long long)N * 48);

        if (ci == 1 || ci == 3) {
            // position update
            g = pos[0];
            k_gvp_vec<<<cdv(N), 256, 0, stream>>>(V, 16, 48, g.Wh, 16, g.Wu, 16, nvu, sbf, 160, 128, N);
            gemm(sbf, 160, g.lW, g.lb, nullptr, 0, ubufB, 160, N, 128, 1);
            gemm(ubufB, 160, g.gW, g.gb, gate, 16, nullptr, 0, N, 16, 2);
            k_gate_mul<<<cdv((long long)N * 16), 256, 0, stream>>>(gate, 16, nvu, nmv, 48, 16, N);
            g = pos[1];
            k_gvp_vec<<<cdv(N), 256, 0, stream>>>(nmv, 16, 48, g.Wh, 16, g.Wu, 16, nvu, ubufB, 160, 128, N);
            gemm(ubufB, 160, g.lW, g.lb, nullptr, 0, ubufC, 160, N, 128, 1);
            gemm(ubufC, 160, g.gW, g.gb, gate, 16, nullptr, 0, N, 16, 2);
            k_gate_mul<<<cdv((long long)N * 16), 256, 0, stream>>>(gate, 16, nvu, nmv, 48, 16, N);
            g = pos[2];
            k_gvp_vec<<<cdv(N), 256, 0, stream>>>(nmv, 16, 48, g.Wh, 16, g.Wu, 1, nvu, ubufC, 160, 128, N);
            gemm(ubufC, 160, g.lW, g.lb, nullptr, 0, ubufB, 160, N, 128, 1);
            gemm(ubufB, 160, g.gW, g.gb, gate, 16, nullptr, 0, N, 1, 0);   // no sigmoid on last gate
            k_gate_mul<<<cdv((long long)N), 256, 0, stream>>>(gate, 16, nvu, xupd, 3, 1, N);
            k_axpy<<<cdv((long long)N * 3), 256, 0, stream>>>(xbuf, xupd, (long long)N * 3);
            // edge update
            k_gather_eupd<<<cdv((long long)E * 384), 256, 0, stream>>>(sbf, efbf, I(210), I(211), h_in, E);
            gemm(h_in, 384, Weu1, beu1, nullptr, 0, hbf, 128, E, 128, 1);
            gemm(hbf, 128, Weu2, beu2, ef_h, 128, nullptr, 0, E, 128, 1);
            k_resid_ln<<<(E + 7) / 8, 256, 0, stream>>>(ef, ef_h, eu_lng, eu_lnb, ef, efbf, 128, E);
            if (ci == 1)
                k_dist<<<cdv(E), 256, 0, stream>>>(xbuf, I(210), I(211), xdiff, lin0, 320, 128, E);
        }
    }

    // ---- heads; out layout: x[30000] | a[100000] | c[60000] | e[400000] ----
    float* out = (float*)d_out;
    gemm(sbf, 160, Wnh1, bnh1, nullptr, 0, nhbf, 128, N, 128, 1);
    gemm(nhbf, 128, Wnh2a, bnh2, out + 30000, 10, nullptr, 0, N, 10, 0);
    gemm(nhbf, 128, Wnh2c, bnh2 + 10, out + 130000, 6, nullptr, 0, N, 6, 0);
    k_copy_f<<<cdv((long long)N * 3), 256, 0, stream>>>(out, xbuf, (long long)N * 3);
    k_add_halves<<<cdv((long long)EH * 128), 256, 0, stream>>>(ef, aebf, EH);
    gemm(aebf, 128, Weh1, beh1, nullptr, 0, ehbf, 128, EH, 128, 1);
    gemm(ehbf, 128, Weh2, beh2, out + 190000, 5, nullptr, 0, EH, 5, 0);
}
